// AttentionBlock_66082366816457
// MI455X (gfx1250) — matrix operand order repaired and checked by reading, not compiled, not run on hardware
//
#include <hip/hip_runtime.h>
#include <hip/hip_bf16.h>

typedef __bf16 bf16_t;
typedef __attribute__((ext_vector_type(16))) __bf16 v16bf;
typedef __attribute__((ext_vector_type(8)))  __bf16 v8bf;
typedef __attribute__((ext_vector_type(8)))  float  v8f;

static constexpr int kBATCH = 16;
static constexpr int kN     = 4096;   // H*W
static constexpr int kC     = 256;
static constexpr int kBN    = kBATCH * kN;   // 65536

#define WMMA_BF16(a, b, c) \
  __builtin_amdgcn_wmma_f32_16x16x32_bf16(false, (a), false, (b), (short)0, (c), false, false)

// Load a 16-element bf16 A/B fragment from two 8-element (16-byte) chunks.
__device__ __forceinline__ v16bf ld8x2(const bf16_t* p0, const bf16_t* p1) {
  v8bf lo = *(const v8bf*)p0;
  v8bf hi = *(const v8bf*)p1;
  v16bf r;
#pragma unroll
  for (int i = 0; i < 8; ++i) { r[i] = lo[i]; r[i + 8] = hi[i]; }
  return r;
}
__device__ __forceinline__ v16bf ld16c(const bf16_t* p) { return ld8x2(p, p + 8); }

// LDS byte address of a __shared__ object (generic ptr low 32 bits = LDS offset).
__device__ __forceinline__ unsigned lds_addr_of(const void* p) {
  return (unsigned)(uintptr_t)p;
}

// Async copy of 64 contiguous bytes global -> LDS per lane (CDNA5 ASYNCcnt path,
// bypasses VGPRs).  INST_OFFSET is applied to both LDS and global addresses.
__device__ __forceinline__ void async_copy64(unsigned lds, const void* g) {
  asm volatile(
      "global_load_async_to_lds_b128 %0, %1, off\n\t"
      "global_load_async_to_lds_b128 %0, %1, off offset:16\n\t"
      "global_load_async_to_lds_b128 %0, %1, off offset:32\n\t"
      "global_load_async_to_lds_b128 %0, %1, off offset:48"
      :: "v"(lds), "v"(g) : "memory");
}
__device__ __forceinline__ void wait_async0() {
  asm volatile("s_wait_asynccnt 0" ::: "memory");
}

// ---------------------------------------------------------------------------
// Kernel 1: convert weights fp32 [C,C] (w[k][n]) -> bf16 transposed wT[n][k].
// 4 weights packed consecutively: wq, wk, wv, wp.
// ---------------------------------------------------------------------------
__global__ void prep_weights_kernel(const float* __restrict__ wq,
                                    const float* __restrict__ wk,
                                    const float* __restrict__ wv,
                                    const float* __restrict__ wp,
                                    bf16_t* __restrict__ wT) {
  int idx = blockIdx.x * 256 + threadIdx.x;      // 0 .. 4*65536-1
  int w   = idx >> 16;
  int n   = (idx >> 8) & 255;
  int k   = idx & 255;
  const float* src = (w == 0) ? wq : (w == 1) ? wk : (w == 2) ? wv : wp;
  wT[idx] = (bf16_t)src[k * kC + n];
}

// ---------------------------------------------------------------------------
// Kernel 2: QKV projections.  grid = (BN/32, 3).  256 threads = 8 waves.
// Block computes a 32-row x 256-col tile of q/k/v; x tile staged bf16 in LDS.
// mat 0/1 -> row-major [BN, C]; mat 2 (V) -> transposed [B, C, N].
// ---------------------------------------------------------------------------
__global__ __launch_bounds__(256) void qkv_kernel(
    const float* __restrict__ x, const bf16_t* __restrict__ wT,
    const float* __restrict__ bq, const float* __restrict__ bk,
    const float* __restrict__ bv,
    bf16_t* __restrict__ qb, bf16_t* __restrict__ kb, bf16_t* __restrict__ vtb) {
  __shared__ __align__(16) bf16_t xs[32 * kC];   // 16 KB

  const int mat  = blockIdx.y;
  const int row0 = blockIdx.x * 32;
  const int t    = threadIdx.x;

  {  // cooperative load x tile (32 x 256 fp32 -> bf16 conversion needs VGPRs)
    int r = t >> 3, seg = t & 7;                 // 8 threads per row, 32 elems each
    const float* src = x + (size_t)(row0 + r) * kC + seg * 32;
    bf16_t* dst = xs + r * kC + seg * 32;
#pragma unroll
    for (int i = 0; i < 8; ++i) {
      float4 f = ((const float4*)src)[i];
      dst[i * 4 + 0] = (bf16_t)f.x; dst[i * 4 + 1] = (bf16_t)f.y;
      dst[i * 4 + 2] = (bf16_t)f.z; dst[i * 4 + 3] = (bf16_t)f.w;
    }
  }
  __syncthreads();

  const int wave = t >> 5, lane = t & 31;
  const int m = lane & 15, hf = lane >> 4;
  const int wrow = (wave & 1) * 16;              // row sub-tile in block
  const int c0   = (wave >> 1) * 64;             // 64-col strip
  const bf16_t* wTm = wT + mat * (kC * kC);

  v8f acc[4] = {};
#pragma unroll
  for (int kk = 0; kk < 8; ++kk) {
    const int kb0 = kk * 32;
    const bf16_t* ap = xs + (wrow + m) * kC + kb0;
    v16bf a = ld16c(ap + 16 * hf);
#pragma unroll
    for (int n = 0; n < 4; ++n) {
      const bf16_t* bp = wTm + (size_t)(c0 + n * 16 + m) * kC + kb0 + 16 * hf;
      v16bf b = ld16c(bp);
      acc[n] = WMMA_BF16(a, b, acc[n]);
    }
  }

  const float* bias = (mat == 0) ? bq : (mat == 1) ? bk : bv;
  const int grow0 = row0 + wrow;
  if (mat < 2) {
    bf16_t* out = (mat == 0) ? qb : kb;
#pragma unroll
    for (int n = 0; n < 4; ++n) {
      int col = c0 + n * 16 + m;
      float bvl = bias[col];
#pragma unroll
      for (int r = 0; r < 8; ++r)
        out[(size_t)(grow0 + r + 8 * hf) * kC + col] = (bf16_t)(acc[n][r] + bvl);
    }
  } else {
    // V transposed: vtb[b][c][tok].  8 consecutive tokens -> one 16B store.
#pragma unroll
    for (int n = 0; n < 4; ++n) {
      int col  = c0 + n * 16 + m;
      float bvl = bias[col];
      int grow = grow0 + 8 * hf;
      int bidx = grow >> 12, tok = grow & (kN - 1);
      bf16_t* dst = vtb + ((size_t)bidx * kC + col) * kN + tok;
      v8bf pk;
#pragma unroll
      for (int r = 0; r < 8; ++r) pk[r] = (bf16_t)(acc[n][r] + bvl);
      *(v8bf*)dst = pk;
    }
  }
}

// ---------------------------------------------------------------------------
// Kernel 3: flash attention.  grid = B * N/128 = 512 blocks, 256 thr = 8 waves.
// Wave owns 16 query rows; block shares double-buffered 32-key K/V LDS tiles
// filled with CDNA5 async global->LDS DMA, overlapped with WMMA compute.
// ---------------------------------------------------------------------------
__global__ __launch_bounds__(256) void attn_kernel(
    const bf16_t* __restrict__ qb, const bf16_t* __restrict__ kb,
    const bf16_t* __restrict__ vtb, bf16_t* __restrict__ ob) {
  __shared__ __align__(16) bf16_t ks_[2][32 * kC];   // K tiles row-major, 2x16 KB
  __shared__ __align__(16) bf16_t vs_[2][kC * 32];   // V^T tiles [c][k],  2x16 KB
  __shared__ __align__(16) bf16_t ps_[8][16 * 32];   // per-wave P,           8 KB

  const int bidx = blockIdx.x >> 5;
  const int qt   = blockIdx.x & 31;
  const int t    = threadIdx.x;
  const int wave = t >> 5, lane = t & 31;
  const int m = lane & 15, hf = lane >> 4;
  const int q0 = qt * 128 + wave * 16;

  const bf16_t* qbase = qb + ((size_t)bidx * kN + q0) * kC;
  const bf16_t* kbase = kb + (size_t)bidx * kN * kC;
  const bf16_t* vbase = vtb + (size_t)bidx * kC * kN;

  // Per-thread async-copy assignments: 64 B of K tile + 64 B of V^T tile.
  const int lr = t >> 3, lseg = t & 7;               // 8 threads per K row
  const bf16_t* kg0 = kbase + (size_t)lr * kC + lseg * 32;
  const bf16_t* vg0 = vbase + (size_t)t * kN;
  unsigned lk[2], lv[2];
#pragma unroll
  for (int bfi = 0; bfi < 2; ++bfi) {
    lk[bfi] = lds_addr_of(&ks_[bfi][lr * kC + lseg * 32]);
    lv[bfi] = lds_addr_of(&vs_[bfi][t * 32]);
  }

  // Preload 8 Q A-fragments (16 x 256 bf16).
  v16bf qf[8];
#pragma unroll
  for (int kk = 0; kk < 8; ++kk) {
    const bf16_t* p = qbase + m * kC + kk * 32;
    qf[kk] = ld16c(p + 16 * hf);
  }

  v8f O[16];
#pragma unroll
  for (int c = 0; c < 16; ++c) O[c] = (v8f){};
  float mrow[8], lrow[8];
#pragma unroll
  for (int r = 0; r < 8; ++r) { mrow[r] = -1e30f; lrow[r] = 0.0f; }
  const float scale = 0.0625f;  // C^-0.5 = 1/16

  // Prologue: kick off tile 0 into buffer 0.
  async_copy64(lk[0], kg0);
  async_copy64(lv[0], vg0);

#pragma unroll 1
  for (int kt = 0; kt < kN / 32; ++kt) {
    const int cur  = kt & 1;
    const int key0 = kt * 32;

    wait_async0();        // this wave's tile-kt DMA landed in LDS
    __syncthreads();      // all waves' DMAs landed; prev compute done

    if (kt + 1 < kN / 32) {   // overlap next tile's DMA with this compute
      async_copy64(lk[cur ^ 1], kg0 + (size_t)(key0 + 32) * kC);
      async_copy64(lv[cur ^ 1], vg0 + key0 + 32);
    }

    const bf16_t* ksb = ks_[cur];
    const bf16_t* vsb = vs_[cur];

    // ---- S = Q K^T for two 16-key sub-tiles ----
    v8f S0 = (v8f){}, S1 = (v8f){};
#pragma unroll
    for (int kk = 0; kk < 8; ++kk) {
      v16bf b0 = ld16c(ksb + m * kC + kk * 32 + 16 * hf);
      v16bf b1 = ld16c(ksb + (16 + m) * kC + kk * 32 + 16 * hf);
      S0 = WMMA_BF16(qf[kk], b0, S0);
      S1 = WMMA_BF16(qf[kk], b1, S1);
    }

    // ---- online softmax (row reductions across 16-lane halves) ----
    float alph[8];
#pragma unroll
    for (int r = 0; r < 8; ++r) {
      float s0 = S0[r] * scale, s1 = S1[r] * scale;
      float mx = fmaxf(s0, s1);
      mx = fmaxf(mx, __shfl_xor(mx, 1, 32));
      mx = fmaxf(mx, __shfl_xor(mx, 2, 32));
      mx = fmaxf(mx, __shfl_xor(mx, 4, 32));
      mx = fmaxf(mx, __shfl_xor(mx, 8, 32));
      float mnew = fmaxf(mrow[r], mx);
      float a    = __expf(mrow[r] - mnew);
      mrow[r] = mnew;
      float p0 = __expf(s0 - mnew), p1 = __expf(s1 - mnew);
      float rs = p0 + p1;
      rs += __shfl_xor(rs, 1, 32);
      rs += __shfl_xor(rs, 2, 32);
      rs += __shfl_xor(rs, 4, 32);
      rs += __shfl_xor(rs, 8, 32);
      lrow[r] = lrow[r] * a + rs;
      alph[r] = a;
      ps_[wave][(r + 8 * hf) * 32 + m]      = (bf16_t)p0;
      ps_[wave][(r + 8 * hf) * 32 + 16 + m] = (bf16_t)p1;
    }
#pragma unroll
    for (int c = 0; c < 16; ++c)
#pragma unroll
      for (int r = 0; r < 8; ++r) O[c][r] *= alph[r];

    // ---- O += P V  (P re-shaped via LDS into A-fragment) ----
    const bf16_t* pp = ps_[wave] + m * 32;
    v16bf pa = ld16c(pp + 16 * hf);
#pragma unroll
    for (int c = 0; c < 16; ++c) {
      v16bf bvf = ld16c(vsb + (c * 16 + m) * 32 + 16 * hf);
      O[c] = WMMA_BF16(pa, bvf, O[c]);
    }
  }

  // ---- normalize and store attention output (bf16, row-major) ----
  bf16_t* out = ob + ((size_t)bidx * kN + q0) * kC;
  float inv[8];
#pragma unroll
  for (int r = 0; r < 8; ++r) inv[r] = 1.0f / lrow[r];
#pragma unroll
  for (int c = 0; c < 16; ++c)
#pragma unroll
    for (int r = 0; r < 8; ++r)
      out[(size_t)(r + 8 * hf) * kC + c * 16 + m] = (bf16_t)(O[c][r] * inv[r]);
}

// ---------------------------------------------------------------------------
// Kernel 4: proj + bias + residual.  grid = BN/32, 256 threads.
// ob tile staged in LDS via async global->LDS DMA (pure bf16 copy).
// ---------------------------------------------------------------------------
__global__ __launch_bounds__(256) void proj_kernel(
    const bf16_t* __restrict__ ob, const bf16_t* __restrict__ wT,
    const float* __restrict__ bp, const float* __restrict__ x,
    float* __restrict__ out) {
  __shared__ __align__(16) bf16_t os[32 * kC];

  const int row0 = blockIdx.x * 32;
  const int t = threadIdx.x;
  {
    int r = t >> 3, seg = t & 7;
    async_copy64(lds_addr_of(&os[r * kC + seg * 32]),
                 ob + (size_t)(row0 + r) * kC + seg * 32);
    wait_async0();
  }
  __syncthreads();

  const int wave = t >> 5, lane = t & 31;
  const int m = lane & 15, hf = lane >> 4;
  const int wrow = (wave & 1) * 16;
  const int c0   = (wave >> 1) * 64;
  const bf16_t* wpT = wT + 3 * (kC * kC);

  v8f acc[4] = {};
#pragma unroll
  for (int kk = 0; kk < 8; ++kk) {
    const int kb0 = kk * 32;
    const bf16_t* ap = os + (wrow + m) * kC + kb0;
    v16bf a = ld16c(ap + 16 * hf);
#pragma unroll
    for (int n = 0; n < 4; ++n) {
      v16bf b = ld16c(wpT + (size_t)(c0 + n * 16 + m) * kC + kb0 + 16 * hf);
      acc[n] = WMMA_BF16(a, b, acc[n]);
    }
  }

  const int grow0 = row0 + wrow;
#pragma unroll
  for (int n = 0; n < 4; ++n) {
    int col = c0 + n * 16 + m;
    float bvl = bp[col];
#pragma unroll
    for (int r = 0; r < 8; ++r) {
      size_t idx = (size_t)(grow0 + r + 8 * hf) * kC + col;
      out[idx] = x[idx] + acc[n][r] + bvl;
    }
  }
}

// ---------------------------------------------------------------------------
extern "C" void kernel_launch(void* const* d_in, const int* in_sizes, int n_in,
                              void* d_out, int out_size, void* d_ws, size_t ws_size,
                              hipStream_t stream) {
  (void)in_sizes; (void)n_in; (void)out_size; (void)ws_size;
  const float* x  = (const float*)d_in[0];
  const float* wq = (const float*)d_in[1];
  const float* bq = (const float*)d_in[2];
  const float* wk = (const float*)d_in[3];
  const float* bk = (const float*)d_in[4];
  const float* wv = (const float*)d_in[5];
  const float* bv = (const float*)d_in[6];
  const float* wp = (const float*)d_in[7];
  const float* bp = (const float*)d_in[8];

  bf16_t* ws  = (bf16_t*)d_ws;
  bf16_t* wT  = ws;                                  // 4 * 256*256
  bf16_t* qb  = wT  + 4 * (size_t)kC * kC;           // BN*C
  bf16_t* kb  = qb  + (size_t)kBN * kC;
  bf16_t* vtb = kb  + (size_t)kBN * kC;              // [B, C, N]
  bf16_t* ob  = vtb + (size_t)kBN * kC;

  prep_weights_kernel<<<4 * kC * kC / 256, 256, 0, stream>>>(wq, wk, wv, wp, wT);
  qkv_kernel<<<dim3(kBN / 32, 3), 256, 0, stream>>>(x, wT, bq, bk, bv, qb, kb, vtb);
  attn_kernel<<<kBATCH * (kN / 128), 256, 0, stream>>>(qb, kb, vtb, ob);
  proj_kernel<<<kBN / 32, 256, 0, stream>>>(ob, wT, bp, x, (float*)d_out);
}
